// DualAttentionHead_32444182954276
// MI455X (gfx1250) — compile-verified, matrix operand order checked by reading, not run on hardware
//
#include <hip/hip_runtime.h>
#include <stdint.h>

// ---------------------------------------------------------------------------
// DANet dual-attention head for MI455X (gfx1250, wave32, WMMA bf16).
// v_wmma_f32_16x16x32_bf16 everywhere; ds_load_b128 fragment feeds;
// ping-pong LDS tiles (1 barrier / K-step); vectorized staging & epilogues;
// im2col tap decode kept in registers (incremental, division-free).
// ---------------------------------------------------------------------------

typedef __bf16 bf16;
typedef __attribute__((ext_vector_type(16))) __bf16 v16bf;
typedef __attribute__((ext_vector_type(8)))  __bf16 v8bf;
typedef __attribute__((ext_vector_type(8)))  float  v8f;

__device__ __forceinline__ float loadVal(const float* p) { return *p; }
__device__ __forceinline__ float loadVal(const bf16* p)  { return (float)*p; }
__device__ __forceinline__ void  storeVal(float* p, float v) { *p = v; }
__device__ __forceinline__ void  storeVal(bf16* p, float v)  { *p = (bf16)v; }

#define WMMA_BF16(A_, B_, C_) \
    __builtin_amdgcn_wmma_f32_16x16x32_bf16(false, (A_), false, (B_), (short)0, (C_), false, false)

__device__ __forceinline__ v16bf cat8(v8bf lo, v8bf hi)
{
    return __builtin_shufflevector(lo, hi, 0, 1, 2, 3, 4, 5, 6, 7,
                                           8, 9, 10, 11, 12, 13, 14, 15);
}

// Fragment load: two aligned 16B LDS reads per ISA 16-bit 16x32 layout.
template <int PITCH>
__device__ __forceinline__ v16bf ldfrag(const bf16 (*T)[PITCH], int row, int hlf)
{
    const v8bf lo = *(const v8bf*)&T[row][8 * hlf];
    const v8bf hi = *(const v8bf*)&T[row][16 + 8 * hlf];
    return cat8(lo, hi);
}

// ---------------------------------------------------------------------------
// Generic batched strided GEMM:  C[b] = A[b] (MxK) * B[b] (KxN)  (+bias, relu)
// Block: 256 threads (8 waves). Tile: 64(M) x 128(N), BK=32, double-buffered.
// Waves: 2(m) x 4(n); each wave computes 32x32 = 4 WMMA accumulators.
// Requirements (all call sites satisfy): K%8==0, N%8==0,
//   aC==1 -> aR%8==0;  bR==1 -> bC%8==0;  bC==1 -> bR%8==0; 16B-aligned bufs.
// ---------------------------------------------------------------------------
template <typename TC>
__global__ __launch_bounds__(256)
void gemm_wmma_bf16(const bf16* __restrict__ A, long aB, long aR, long aC,
                    const bf16* __restrict__ Bm, long bB, long bR, long bC,
                    TC* __restrict__ Cm, long cB, long cR, long cC,
                    const float* __restrict__ bias, int biasMode, int doRelu,
                    int M, int N, int K)
{
    __shared__ __align__(16) bf16 As[2][64][40];    // m-major
    __shared__ __align__(16) bf16 Bs[2][128][40];   // n-major

    const bf16* Ab = A  + (long)blockIdx.z * aB;
    const bf16* Bb = Bm + (long)blockIdx.z * bB;
    TC*         Cb = Cm + (long)blockIdx.z * cB;

    const int m0   = blockIdx.y * 64;
    const int n0   = blockIdx.x * 128;
    const int tid  = threadIdx.x;
    const int wave = tid >> 5;
    const int lane = tid & 31;
    const int wm   = (wave >> 2) * 32;   // 0,32
    const int wn   = (wave & 3) * 32;    // 0..96
    const int ln   = lane & 15;
    const int hlf  = lane >> 4;

    const v8f zz = {};
    v8f acc00 = zz, acc01 = zz, acc10 = zz, acc11 = zz;

    auto stageA = [&](int buf, int k0) {
        if (aC == 1) {
            // K-contiguous: one 16B chunk per thread (64*32/8 == 256)
            int r = tid >> 2, c8 = (tid & 3) * 8;
            int gm = m0 + r, gk = k0 + c8;
            v8bf v = {};
            if (gm < M && gk < K) v = *(const v8bf*)&Ab[(long)gm * aR + gk];
            *(v8bf*)&As[buf][r][c8] = v;
        } else {
            // M-contiguous (transposed A): scalar, lanes sweep m for coalescing
            for (int t = tid; t < 2048; t += 256) {
                int r = t & 63, c = t >> 6;
                int gm = m0 + r, gk = k0 + c;
                float v = 0.f;
                if (gm < M && gk < K) v = (float)Ab[(long)gm * aR + (long)gk * aC];
                As[buf][r][c] = (bf16)v;
            }
        }
    };
    auto stageB = [&](int buf, int k0) {
        if (bR == 1) {
            // K-contiguous: fully vectorized (128*32/8 == 512 chunks)
            for (int t = tid; t < 512; t += 256) {
                int n = t >> 2, k8 = (t & 3) * 8;
                int gn = n0 + n, gk = k0 + k8;
                v8bf v = {};
                if (gn < N && gk < K) v = *(const v8bf*)&Bb[(long)gn * bC + gk];
                *(v8bf*)&Bs[buf][n][k8] = v;
            }
        } else {
            // N-contiguous: vector global load, transpose-scatter into LDS
            for (int t = tid; t < 512; t += 256) {
                int kk = t >> 4, n8 = (t & 15) * 8;
                int gk = k0 + kk, gn = n0 + n8;
                v8bf v = {};
                if (gk < K && gn < N) v = *(const v8bf*)&Bb[(long)gk * bR + (long)gn * bC];
#pragma unroll
                for (int j = 0; j < 8; ++j) Bs[buf][n8 + j][kk] = v[j];
            }
        }
    };

    const int nT = (K + 31) >> 5;
    stageA(0, 0);
    stageB(0, 0);
    __syncthreads();

    for (int i = 0; i < nT; ++i) {
        const int cur = i & 1, nxt = cur ^ 1;
        if (i + 1 < nT) {               // overlap next-tile loads with WMMA
            stageA(nxt, (i + 1) * 32);
            stageB(nxt, (i + 1) * 32);
        }
        if (i + 2 < nT && tid < 64) {   // prefetch tile i+2 (global_prefetch_b8)
            int k2 = (i + 2) * 32;
            if (tid < 32) {
                int gm = m0 + tid * 2;
                if (gm < M) __builtin_prefetch(&Ab[(long)gm * aR + (long)k2 * aC], 0, 1);
            } else {
                int gk = k2 + (tid - 32);
                if (gk < K) __builtin_prefetch(&Bb[(long)gk * bR + (long)n0 * bC], 0, 1);
            }
        }
        v16bf af0 = ldfrag<40>(As[cur], wm + ln, hlf);
        v16bf af1 = ldfrag<40>(As[cur], wm + 16 + ln, hlf);
        v16bf bf0 = ldfrag<40>(Bs[cur], wn + ln, hlf);
        v16bf bf1 = ldfrag<40>(Bs[cur], wn + 16 + ln, hlf);
        acc00 = WMMA_BF16(af0, bf0, acc00);
        acc01 = WMMA_BF16(af0, bf1, acc01);
        acc10 = WMMA_BF16(af1, bf0, acc10);
        acc11 = WMMA_BF16(af1, bf1, acc11);
        __syncthreads();                // one barrier per K-step
    }

    // ---- epilogue: D layout m = r + 8*(lane>=16), n = lane&15 ----
    const int mh = hlf * 8;
#pragma unroll
    for (int r = 0; r < 8; ++r) {
        int gn0 = n0 + wn + ln;
        int gn1 = gn0 + 16;
        int gmA = m0 + wm + mh + r;
        int gmB = gmA + 16;
        if (gmA < M) {
            float bv = (biasMode == 1) ? bias[gmA] : 0.f;
            float v0 = acc00[r] + bv, v1 = acc01[r] + bv;
            if (doRelu) { v0 = fmaxf(v0, 0.f); v1 = fmaxf(v1, 0.f); }
            if (gn0 < N) storeVal(&Cb[(long)gmA * cR + (long)gn0 * cC], v0);
            if (gn1 < N) storeVal(&Cb[(long)gmA * cR + (long)gn1 * cC], v1);
        }
        if (gmB < M) {
            float bv = (biasMode == 1) ? bias[gmB] : 0.f;
            float v0 = acc10[r] + bv, v1 = acc11[r] + bv;
            if (doRelu) { v0 = fmaxf(v0, 0.f); v1 = fmaxf(v1, 0.f); }
            if (gn0 < N) storeVal(&Cb[(long)gmB * cR + (long)gn0 * cC], v0);
            if (gn1 < N) storeVal(&Cb[(long)gmB * cR + (long)gn1 * cC], v1);
        }
    }
}

// ---------------------------------------------------------------------------
// Implicit-GEMM 3x3 conv (pad=1, H=W=64), NCHW, BN folded, fused ReLU,
// dual f32+bf16 output. rows = b*4096+pixel, cols = CO (==128, grid.x==1),
// K = Cin*9 (multiple of 32). Weights wT are [CO][K] bf16.
// Im2col tap decode (channel, ky, kx) lives in per-thread registers and is
// advanced incrementally per tile (32 = 3*9 + 5): no division, no LDS tables,
// no exec-mask branches in the hot loop (clamped address + select).
// ---------------------------------------------------------------------------
template <typename TIN>
__global__ __launch_bounds__(256)
void conv3x3_wmma(const TIN* __restrict__ in, int Cin,
                  const bf16* __restrict__ wT, int CO,
                  const float* __restrict__ bias,
                  float* __restrict__ outF, bf16* __restrict__ outB,
                  int Mtot)
{
    __shared__ __align__(16) bf16 As[2][64][40];
    __shared__ __align__(16) bf16 Bs[2][128][40];

    const int K    = Cin * 9;            // multiple of 32 for Cin in {512,128}
    const int nT   = K >> 5;
    const int m0   = blockIdx.y * 64;
    const int tid  = threadIdx.x;
    const int wave = tid >> 5;
    const int lane = tid & 31;
    const int wm   = (wave >> 2) * 32;
    const int wn   = (wave & 3) * 32;
    const int ln   = lane & 15;
    const int hlf  = lane >> 4;

    const int  bimg = m0 >> 12;          // whole 64-row block is one batch img
    const int  p0   = m0 & 4095;
    const long ibase = ((long)bimg * Cin) << 12;

    // This thread's staged pixel (row) and k-columns (c0 + 4*t2, t2=0..7).
    const int rpix = tid & 63;
    const int c0   = tid >> 6;
    const int py   = (p0 + rpix) >> 6;   // pixel y
    const int px   = (p0 + rpix) & 63;   // pixel x

    // Register-resident tap decode state for each of the 8 k-columns.
    int cis[8], tts[8];
#pragma unroll
    for (int t2 = 0; t2 < 8; ++t2) {
        int c = c0 + t2 * 4;             // one-time small div (compile-time-ish)
        cis[t2] = c / 9;
        tts[t2] = c - cis[t2] * 9;
    }

    const v8f zz = {};
    v8f acc00 = zz, acc01 = zz, acc10 = zz, acc11 = zz;

    auto stageA = [&](int buf) {         // im2col from register tap state
#pragma unroll
        for (int t2 = 0; t2 < 8; ++t2) {
            int tt = tts[t2];
            int t3 = tt / 3;             // tt in [0,9): cheap mul-shift
            int ky = t3 - 1;
            int kx = (tt - t3 * 3) - 1;
            int y  = py + ky;
            int x  = px + kx;
            bool inb = ((unsigned)y < 64u) && ((unsigned)x < 64u);
            int yc = y < 0 ? 0 : (y > 63 ? 63 : y);   // clamp: load always legal
            int xc = x < 0 ? 0 : (x > 63 ? 63 : x);
            long idx = ibase + ((long)cis[t2] << 12) + (yc << 6) + xc;
            float vv = loadVal(&in[idx]);
            As[buf][rpix][c0 + t2 * 4] = (bf16)(inb ? vv : 0.f);
        }
    };
    auto advTaps = [&]() {               // advance decode by +32 k
#pragma unroll
        for (int t2 = 0; t2 < 8; ++t2) {
            cis[t2] += 3;
            tts[t2] += 5;
            if (tts[t2] >= 9) { tts[t2] -= 9; cis[t2] += 1; }
        }
    };
    auto stageB = [&](int buf, int k0) {
#pragma unroll
        for (int t2 = 0; t2 < 2; ++t2) {
            int t = tid + t2 * 256;
            int n = t >> 2, k8 = (t & 3) * 8;
            v8bf v = {};
            if (n < CO) v = *(const v8bf*)&wT[(long)n * K + k0 + k8];
            *(v8bf*)&Bs[buf][n][k8] = v;
        }
    };

    stageA(0);
    stageB(0, 0);
    advTaps();                           // state now describes tile 1
    __syncthreads();

    for (int i = 0; i < nT; ++i) {
        const int cur = i & 1, nxt = cur ^ 1;
        if (i + 1 < nT) {
            stageA(nxt);
            stageB(nxt, (i + 1) * 32);
            advTaps();                   // state -> tile i+2
        }
        if (i + 2 < nT && tid >= 128 && tid < 192)
            __builtin_prefetch(&wT[(long)(tid - 128) * K + (i + 2) * 32], 0, 1);

        v16bf af0 = ldfrag<40>(As[cur], wm + ln, hlf);
        v16bf af1 = ldfrag<40>(As[cur], wm + 16 + ln, hlf);
        v16bf bf0 = ldfrag<40>(Bs[cur], wn + ln, hlf);
        v16bf bf1 = ldfrag<40>(Bs[cur], wn + 16 + ln, hlf);
        acc00 = WMMA_BF16(af0, bf0, acc00);
        acc01 = WMMA_BF16(af0, bf1, acc01);
        acc10 = WMMA_BF16(af1, bf0, acc10);
        acc11 = WMMA_BF16(af1, bf1, acc11);
        __syncthreads();
    }

    // ---- epilogue: each lane owns 8 consecutive pixels -> vector stores ----
    const int mh = hlf * 8;
#pragma unroll
    for (int h2 = 0; h2 < 2; ++h2) {
        int gm0 = m0 + wm + mh + 16 * h2;          // rows gm0..gm0+7 (same b)
        int b = gm0 >> 12, p = gm0 & 4095;
#pragma unroll
        for (int nn = 0; nn < 2; ++nn) {
            int gn = wn + ln + 16 * nn;
            if (gn >= CO) continue;
            float bv = bias[gn];
            float tmp[8];
            v8bf  ob;
#pragma unroll
            for (int r = 0; r < 8; ++r) {
                float v;
                if (h2 == 0) v = nn == 0 ? acc00[r] : acc01[r];
                else         v = nn == 0 ? acc10[r] : acc11[r];
                v = fmaxf(v + bv, 0.f);
                tmp[r] = v;
                ob[r]  = (bf16)v;
            }
            long idx = (((long)b * CO + gn) << 12) + p;   // 8-aligned
            float4 fa = {tmp[0], tmp[1], tmp[2], tmp[3]};
            float4 fb = {tmp[4], tmp[5], tmp[6], tmp[7]};
            *(float4*)&outF[idx]     = fa;
            *(float4*)&outF[idx + 4] = fb;
            *(v8bf*)&outB[idx]       = ob;
        }
    }
}

// ---------------------------------------------------------------------------
// Row softmax, bf16 in/out, len%8==0, len<=4096; row cached in registers:
// exactly 1 global read + 1 global write (v8bf vector accesses). In-place OK.
// ---------------------------------------------------------------------------
__global__ __launch_bounds__(256)
void softmax_rows_bf16(const bf16* __restrict__ in, bf16* __restrict__ out, int len)
{
    __shared__ float red[256];
    const long row = blockIdx.x;
    const bf16* ip = in  + row * (long)len;
    bf16*       op = out + row * (long)len;
    const int tid = threadIdx.x;

    float lv[16];
    int   pos[2];
    int   nc = 0;
    for (int i = tid * 8; i < len && nc < 2; i += 2048) {
        v8bf v = *(const v8bf*)&ip[i];
#pragma unroll
        for (int j = 0; j < 8; ++j) lv[nc * 8 + j] = (float)v[j];
        pos[nc++] = i;
    }

    float lmax = -3.402823466e38f;
    for (int c = 0; c < nc; ++c)
#pragma unroll
        for (int j = 0; j < 8; ++j) lmax = fmaxf(lmax, lv[c * 8 + j]);
    red[tid] = lmax; __syncthreads();
    for (int s = 128; s > 0; s >>= 1) {
        if (tid < s) red[tid] = fmaxf(red[tid], red[tid + s]);
        __syncthreads();
    }
    const float m = red[0]; __syncthreads();

    float lsum = 0.f;
    for (int c = 0; c < nc; ++c)
#pragma unroll
        for (int j = 0; j < 8; ++j) {
            float e = __expf(lv[c * 8 + j] - m);
            lv[c * 8 + j] = e;
            lsum += e;
        }
    red[tid] = lsum; __syncthreads();
    for (int s = 128; s > 0; s >>= 1) {
        if (tid < s) red[tid] += red[tid + s];
        __syncthreads();
    }
    const float inv = 1.0f / red[0];

    for (int c = 0; c < nc; ++c) {
        v8bf o;
#pragma unroll
        for (int j = 0; j < 8; ++j) o[j] = (bf16)(lv[c * 8 + j] * inv);
        *(v8bf*)&op[pos[c]] = o;
    }
}

// Small-row softmax, f32 in -> bf16 out (CAM: 512 rows of 128).
__global__ __launch_bounds__(256)
void softmax_rows_f32(const float* __restrict__ in, bf16* __restrict__ out, int len)
{
    __shared__ float red[256];
    const long row = blockIdx.x;
    const float* ip = in  + row * (long)len;
    bf16*        op = out + row * (long)len;
    const int tid = threadIdx.x;

    float lmax = -3.402823466e38f;
    for (int i = tid; i < len; i += 256) lmax = fmaxf(lmax, ip[i]);
    red[tid] = lmax; __syncthreads();
    for (int s = 128; s > 0; s >>= 1) {
        if (tid < s) red[tid] = fmaxf(red[tid], red[tid + s]);
        __syncthreads();
    }
    const float m = red[0]; __syncthreads();

    float lsum = 0.f;
    for (int i = tid; i < len; i += 256) lsum += __expf(ip[i] - m);
    red[tid] = lsum; __syncthreads();
    for (int s = 128; s > 0; s >>= 1) {
        if (tid < s) red[tid] += red[tid + s];
        __syncthreads();
    }
    const float inv = 1.0f / red[0];

    for (int i = tid; i < len; i += 256)
        op[i] = (bf16)(__expf(ip[i] - m) * inv);
}

// ---------------------------------------------------------------------------
// Small prep / elementwise kernels
// ---------------------------------------------------------------------------
__global__ __launch_bounds__(256)
void prep_conv_w_kernel(const float* __restrict__ w, const float* __restrict__ g,
                        const float* __restrict__ var, bf16* __restrict__ wT,
                        int CO, int Kk)
{
    long total = (long)CO * Kk;
    for (long i = (long)blockIdx.x * 256 + threadIdx.x; i < total;
         i += (long)gridDim.x * 256) {
        int co = (int)(i / Kk);
        float inv = g[co] * rsqrtf(var[co] + 1e-5f);
        wT[i] = (bf16)(w[i] * inv);
    }
}

__global__ __launch_bounds__(256)
void prep_bn_bias_kernel(const float* g, const float* b, const float* m,
                         const float* v, float* bias, int CO)
{
    int i = threadIdx.x;
    if (i < CO) {
        float inv = g[i] * rsqrtf(v[i] + 1e-5f);
        bias[i] = b[i] - m[i] * inv;
    }
}

__global__ __launch_bounds__(256)
void f32_to_bf16_kernel(const float* __restrict__ a, bf16* __restrict__ o, long n)
{
    long i = (long)blockIdx.x * 256 + threadIdx.x;
    if (i < n) o[i] = (bf16)a[i];
}

__global__ __launch_bounds__(256)
void copy_f32_kernel(const float* __restrict__ a, float* __restrict__ o, int n)
{
    int i = blockIdx.x * 256 + threadIdx.x;
    if (i < n) o[i] = a[i];
}

__global__ __launch_bounds__(256)
void scale_add_bf16_kernel(const float* __restrict__ a, const float* __restrict__ gamma,
                           const float* __restrict__ b, bf16* __restrict__ o, long n)
{
    long i = (long)blockIdx.x * 256 + threadIdx.x;
    if (i < n) o[i] = (bf16)(gamma[0] * a[i] + b[i]);
}

__global__ __launch_bounds__(256)
void add_bf16_kernel(const float* __restrict__ a, const float* __restrict__ b,
                     bf16* __restrict__ o, long n)
{
    long i = (long)blockIdx.x * 256 + threadIdx.x;
    if (i < n) o[i] = (bf16)(a[i] + b[i]);
}

// ---------------------------------------------------------------------------
// Host orchestration
// ---------------------------------------------------------------------------
extern "C" void kernel_launch(void* const* d_in, const int* in_sizes, int n_in,
                              void* d_out, int out_size, void* d_ws, size_t ws_size,
                              hipStream_t stream)
{
    (void)in_sizes; (void)n_in; (void)out_size; (void)ws_size;

    constexpr int  Bt  = 4, CIN = 512, CI = 128, CQK = 16, NC = 19;
    constexpr int  CQKV = CQK + CQK + CI;      // 160 stacked q/k/v rows
    constexpr long Np  = 4096;                 // H*W
    constexpr long FEAT = (long)Bt * CI * Np;  // 2,097,152 elements

    const float* x      = (const float*)d_in[0];
    const float* wp1    = (const float*)d_in[1];
    const float* bnp1_w = (const float*)d_in[2];
    const float* bnp1_b = (const float*)d_in[3];
    const float* bnp1_m = (const float*)d_in[4];
    const float* bnp1_v = (const float*)d_in[5];
    const float* qw     = (const float*)d_in[6];
    const float* qb     = (const float*)d_in[7];
    const float* kw     = (const float*)d_in[8];
    const float* kb     = (const float*)d_in[9];
    const float* vw     = (const float*)d_in[10];
    const float* vb     = (const float*)d_in[11];
    const float* pamg   = (const float*)d_in[12];
    const float* wp2    = (const float*)d_in[13];
    const float* bnp2_w = (const float*)d_in[14];
    const float* bnp2_b = (const float*)d_in[15];
    const float* bnp2_m = (const float*)d_in[16];
    const float* bnp2_v = (const float*)d_in[17];
    const float* wc1    = (const float*)d_in[18];
    const float* bnc1_w = (const float*)d_in[19];
    const float* bnc1_b = (const float*)d_in[20];
    const float* bnc1_m = (const float*)d_in[21];
    const float* bnc1_v = (const float*)d_in[22];
    const float* camg   = (const float*)d_in[23];
    const float* wc2    = (const float*)d_in[24];
    const float* bnc2_w = (const float*)d_in[25];
    const float* bnc2_b = (const float*)d_in[26];
    const float* bnc2_m = (const float*)d_in[27];
    const float* bnc2_v = (const float*)d_in[28];
    const float* cls_w  = (const float*)d_in[29];
    const float* cls_b  = (const float*)d_in[30];
    const float* auxp_w = (const float*)d_in[31];
    const float* auxp_b = (const float*)d_in[32];
    const float* auxc_w = (const float*)d_in[33];
    const float* auxc_b = (const float*)d_in[34];

    float* out_main = (float*)d_out;                        // [4,19,4096]
    float* out_p    = out_main + (long)Bt * NC * Np;
    float* out_c    = out_p    + (long)Bt * NC * Np;

    // ---- workspace bump allocator (~215 MB total) ----
    char*  base = (char*)d_ws;
    size_t off  = 0;
    auto alloc = [&](size_t bytes) -> void* {
        void* p = base + off;
        off = (off + bytes + 255) & ~(size_t)255;
        return p;
    };
    bf16*  wp1T      = (bf16*)alloc((size_t)CIN * 9 * CI * 2);
    bf16*  wc1T      = (bf16*)alloc((size_t)CIN * 9 * CI * 2);
    bf16*  wp2T      = (bf16*)alloc((size_t)CI * 9 * CI * 2);
    bf16*  wc2T      = (bf16*)alloc((size_t)CI * 9 * CI * 2);
    float* bias_p1   = (float*)alloc(CI * 4);
    float* bias_p2   = (float*)alloc(CI * 4);
    float* bias_c1   = (float*)alloc(CI * 4);
    float* bias_c2   = (float*)alloc(CI * 4);
    bf16*  qkvw_bf   = (bf16*)alloc((size_t)CQKV * CI * 2);
    float* qkvb      = (float*)alloc((size_t)CQKV * 4);
    bf16*  clsw_bf   = (bf16*)alloc((size_t)NC * CI * 2);
    bf16*  auxpw_bf  = (bf16*)alloc((size_t)NC * CI * 2);
    bf16*  auxcw_bf  = (bf16*)alloc((size_t)NC * CI * 2);
    float* featp_f   = (float*)alloc(FEAT * 4);
    bf16*  featp_b   = (bf16*)alloc(FEAT * 2);
    float* featc_f   = (float*)alloc(FEAT * 4);
    bf16*  featc_b   = (bf16*)alloc(FEAT * 2);
    bf16*  qkv_bf    = (bf16*)alloc((size_t)Bt * CQKV * Np * 2);
    bf16*  attn      = (bf16*)alloc((size_t)Bt * Np * Np * 2);   // 128 MB
    float* pam_out   = (float*)alloc(FEAT * 4);
    bf16*  p2in_b    = (bf16*)alloc(FEAT * 2);
    float* featp2_f  = (float*)alloc(FEAT * 4);
    bf16*  featp2_b  = (bf16*)alloc(FEAT * 2);
    float* cam_en    = (float*)alloc((size_t)Bt * CI * CI * 4);
    bf16*  cam_attn  = (bf16*)alloc((size_t)Bt * CI * CI * 2);
    float* cam_out   = (float*)alloc(FEAT * 4);
    bf16*  c2in_b    = (bf16*)alloc(FEAT * 2);
    float* featc2_f  = (float*)alloc(FEAT * 4);
    bf16*  featc2_b  = (bf16*)alloc(FEAT * 2);
    bf16*  feat_b    = (bf16*)alloc(FEAT * 2);

    const dim3 blk(256);
    auto ceilDiv = [](long a, long b) { return (unsigned)((a + b - 1) / b); };

    // ---- weight prep (BN folded into conv weights; layout stays [CO][K]) ----
    prep_conv_w_kernel<<<2304, blk, 0, stream>>>(wp1, bnp1_w, bnp1_v, wp1T, CI, CIN * 9);
    prep_conv_w_kernel<<<2304, blk, 0, stream>>>(wc1, bnc1_w, bnc1_v, wc1T, CI, CIN * 9);
    prep_conv_w_kernel<<<576,  blk, 0, stream>>>(wp2, bnp2_w, bnp2_v, wp2T, CI, CI * 9);
    prep_conv_w_kernel<<<576,  blk, 0, stream>>>(wc2, bnc2_w, bnc2_v, wc2T, CI, CI * 9);
    prep_bn_bias_kernel<<<1, blk, 0, stream>>>(bnp1_w, bnp1_b, bnp1_m, bnp1_v, bias_p1, CI);
    prep_bn_bias_kernel<<<1, blk, 0, stream>>>(bnp2_w, bnp2_b, bnp2_m, bnp2_v, bias_p2, CI);
    prep_bn_bias_kernel<<<1, blk, 0, stream>>>(bnc1_w, bnc1_b, bnc1_m, bnc1_v, bias_c1, CI);
    prep_bn_bias_kernel<<<1, blk, 0, stream>>>(bnc2_w, bnc2_b, bnc2_m, bnc2_v, bias_c2, CI);
    // stacked q/k/v weights [160][128] and bias [160]
    f32_to_bf16_kernel<<<ceilDiv(CQK * CI, 256), blk, 0, stream>>>(qw, qkvw_bf, CQK * CI);
    f32_to_bf16_kernel<<<ceilDiv(CQK * CI, 256), blk, 0, stream>>>(kw, qkvw_bf + CQK * CI, CQK * CI);
    f32_to_bf16_kernel<<<ceilDiv(CI * CI, 256), blk, 0, stream>>>(vw, qkvw_bf + 2 * CQK * CI, CI * CI);
    copy_f32_kernel<<<1, blk, 0, stream>>>(qb, qkvb, CQK);
    copy_f32_kernel<<<1, blk, 0, stream>>>(kb, qkvb + CQK, CQK);
    copy_f32_kernel<<<1, blk, 0, stream>>>(vb, qkvb + 2 * CQK, CI);
    f32_to_bf16_kernel<<<ceilDiv(NC * CI, 256), blk, 0, stream>>>(cls_w, clsw_bf, NC * CI);
    f32_to_bf16_kernel<<<ceilDiv(NC * CI, 256), blk, 0, stream>>>(auxp_w, auxpw_bf, NC * CI);
    f32_to_bf16_kernel<<<ceilDiv(NC * CI, 256), blk, 0, stream>>>(auxc_w, auxcw_bf, NC * CI);

    // ---- conv_p1 / conv_c1 (512->128, BN+ReLU fused) ----
    conv3x3_wmma<float><<<dim3(1, Bt * Np / 64), blk, 0, stream>>>(
        x, CIN, wp1T, CI, bias_p1, featp_f, featp_b, Bt * (int)Np);
    conv3x3_wmma<float><<<dim3(1, Bt * Np / 64), blk, 0, stream>>>(
        x, CIN, wc1T, CI, bias_c1, featc_f, featc_b, Bt * (int)Np);

    // ---- PAM: fused q/k/v 1x1 conv:  qkv[b] (160x4096) ----
    gemm_wmma_bf16<bf16><<<dim3(32, 3, Bt), blk, 0, stream>>>(
        qkvw_bf, 0, CI, 1, featp_b, CI * Np, Np, 1,
        qkv_bf, (long)CQKV * Np, Np, 1, qkvb, 1, 0, CQKV, (int)Np, CI);

    // energy[b] (4096x4096) = q[b]^T (N x 16) * k[b] (16 x N); q rows 0-15, k rows 16-31
    gemm_wmma_bf16<bf16><<<dim3(32, 64, Bt), blk, 0, stream>>>(
        qkv_bf, (long)CQKV * Np, 1, Np, qkv_bf + CQK * Np, (long)CQKV * Np, Np, 1,
        attn, Np * Np, Np, 1, nullptr, 0, 0, (int)Np, (int)Np, CQK);

    // softmax over rows, in place (bf16 -> bf16), 1 read + 1 write per element
    softmax_rows_bf16<<<Bt * (unsigned)Np, blk, 0, stream>>>(attn, attn, (int)Np);

    // pam_out[b] (128x4096): O[c,n] = sum_m v[c,m] * attn[n,m]; v rows 32-159
    gemm_wmma_bf16<float><<<dim3(32, 2, Bt), blk, 0, stream>>>(
        qkv_bf + 2 * CQK * Np, (long)CQKV * Np, Np, 1, attn, Np * Np, 1, Np,
        pam_out, CI * Np, Np, 1, nullptr, 0, 0, CI, (int)Np, (int)Np);

    // gamma*out + x  -> bf16 input for conv_p2
    scale_add_bf16_kernel<<<ceilDiv(FEAT, 256), blk, 0, stream>>>(
        pam_out, pamg, featp_f, p2in_b, FEAT);

    // conv_p2 (128->128, BN+ReLU)
    conv3x3_wmma<bf16><<<dim3(1, Bt * Np / 64), blk, 0, stream>>>(
        p2in_b, CI, wp2T, CI, bias_p2, featp2_f, featp2_b, Bt * (int)Np);

    // ---- CAM ----
    gemm_wmma_bf16<float><<<dim3(1, 2, Bt), blk, 0, stream>>>(
        featc_b, CI * Np, Np, 1, featc_b, CI * Np, 1, Np,
        cam_en, (long)CI * CI, CI, 1, nullptr, 0, 0, CI, CI, (int)Np);
    softmax_rows_f32<<<Bt * CI, blk, 0, stream>>>(cam_en, cam_attn, CI);
    gemm_wmma_bf16<float><<<dim3(32, 2, Bt), blk, 0, stream>>>(
        cam_attn, (long)CI * CI, CI, 1, featc_b, CI * Np, Np, 1,
        cam_out, CI * Np, Np, 1, nullptr, 0, 0, CI, (int)Np, CI);
    scale_add_bf16_kernel<<<ceilDiv(FEAT, 256), blk, 0, stream>>>(
        cam_out, camg, featc_f, c2in_b, FEAT);
    conv3x3_wmma<bf16><<<dim3(1, Bt * Np / 64), blk, 0, stream>>>(
        c2in_b, CI, wc2T, CI, bias_c2, featc2_f, featc2_b, Bt * (int)Np);

    // ---- fuse + classifier heads (1x1 convs, 19 classes) ----
    add_bf16_kernel<<<ceilDiv(FEAT, 256), blk, 0, stream>>>(
        featp2_f, featc2_f, feat_b, FEAT);

    gemm_wmma_bf16<float><<<dim3(32, 1, Bt), blk, 0, stream>>>(
        clsw_bf, 0, CI, 1, feat_b, CI * Np, Np, 1,
        out_main, (long)NC * Np, Np, 1, cls_b, 1, 0, NC, (int)Np, CI);
    gemm_wmma_bf16<float><<<dim3(32, 1, Bt), blk, 0, stream>>>(
        auxpw_bf, 0, CI, 1, featp2_b, CI * Np, Np, 1,
        out_p, (long)NC * Np, Np, 1, auxp_b, 1, 0, NC, (int)Np, CI);
    gemm_wmma_bf16<float><<<dim3(32, 1, Bt), blk, 0, stream>>>(
        auxcw_bf, 0, CI, 1, featc2_b, CI * Np, Np, 1,
        out_c, (long)NC * Np, Np, 1, auxc_b, 1, 0, NC, (int)Np, CI);
}
